// PerceiverAttention_52003464020056
// MI455X (gfx1250) — matrix operand order repaired and checked by reading, not compiled, not run on hardware
//
#include <hip/hip_runtime.h>

// ---------------------------------------------------------------------------
// Types / WMMA helpers (CDNA5 wave32, v_wmma_f32_16x16x32_f16)
// ---------------------------------------------------------------------------
typedef __attribute__((ext_vector_type(8)))  _Float16 v8h;
typedef __attribute__((ext_vector_type(16))) _Float16 v16h;
typedef __attribute__((ext_vector_type(8)))  float    v8f;

#define DEV __device__ __forceinline__

// A fragment, 16x32 f16. Lane L(0..15): row M=L, K={0..7,16..23};
// lanes 16..31: same rows, K={8..15,24..31}. base = row-major [M][K], ld halves
// (ld must be a multiple of 8 so the 16B vector loads stay aligned).
DEV v16h frag_a_ld(const _Float16* base, int ld) {
  const int lane = threadIdx.x & 31;
  const _Float16* p = base + (lane & 15) * ld + ((lane >> 4) << 3);
  v8h lo = *(const v8h*)(p);
  v8h hi = *(const v8h*)(p + 16);
  return __builtin_shufflevector(lo, hi, 0,1,2,3,4,5,6,7,8,9,10,11,12,13,14,15);
}

// B fragment, 32x16 f16, read from LDS laid out transposed as [N][K].
// Lane L(0..15): col N=L, K={0..7,16..23}; lanes 16..31: K={8..15,24..31}.
DEV v16h frag_b_ld(const _Float16* base, int ld) {
  const int lane = threadIdx.x & 31;
  const _Float16* p = base + (lane & 15) * ld + ((lane >> 4) << 3);
  v8h lo = *(const v8h*)(p);
  v8h hi = *(const v8h*)(p + 16);
  return __builtin_shufflevector(lo, hi, 0,1,2,3,4,5,6,7,8,9,10,11,12,13,14,15);
}

DEV v8f wmma16(v16h a, v16h b, v8f c) {
  return __builtin_amdgcn_wmma_f32_16x16x32_f16(false, a, false, b, (short)0, c,
                                                false, false);
}

// C/D layout: element r of lane L -> row M = ((L>>4)<<3) + r, col N = L&15.

// CDNA5 async global->LDS copy (16B per lane, tracked by ASYNCcnt, no VGPR
// data): lds_addr = low 32 bits of the flat shared address (LDS aperture puts
// the byte offset in addr[31:0]).
DEV void async_g2l_b128(uint32_t lds_addr, const void* gaddr) {
  asm volatile("global_load_async_to_lds_b128 %0, %1, off"
               :
               : "v"(lds_addr), "v"(gaddr)
               : "memory");
}
DEV void wait_async_all() { asm volatile("s_wait_asynccnt 0x0" ::: "memory"); }
DEV uint32_t lds_off(const void* p) { return (uint32_t)(uintptr_t)p; }

// ---------------------------------------------------------------------------
// Problem constants
// ---------------------------------------------------------------------------
#define N_BT     64       // b*t
#define N_MED    1024
#define N_LAT    64
#define N_J      1088     // media + latents
#define DIMC     1024
#define INNERC   512
#define HEADSC   8
#define DHC      64

// ---------------------------------------------------------------------------
// Kernel 1: fused LayerNorm (x rows then latent rows) -> f16 nin[bt][1088][1024]
// ---------------------------------------------------------------------------
DEV float block_reduce_sum(float v, float* red) {
  const int tid = threadIdx.x;
  red[tid] = v;
  __syncthreads();
#pragma unroll
  for (int s = 128; s > 0; s >>= 1) {
    if (tid < s) red[tid] += red[tid + s];
    __syncthreads();
  }
  float r = red[0];
  __syncthreads();
  return r;
}

__global__ __launch_bounds__(256) void ln_kernel(
    const float* __restrict__ x, const float* __restrict__ lat,
    const float* __restrict__ wm, const float* __restrict__ bm,
    const float* __restrict__ wl, const float* __restrict__ bl,
    _Float16* __restrict__ nin) {
  __shared__ float red[256];
  const int row = blockIdx.x;
  const int tid = threadIdx.x;

  const float* src; const float* w; const float* b; size_t drow;
  if (row < N_BT * N_MED) {
    src = x + (size_t)row * DIMC; w = wm; b = bm;
    drow = (size_t)(row >> 10) * N_J + (row & 1023);
  } else {
    const int r2 = row - N_BT * N_MED;
    src = lat + (size_t)r2 * DIMC; w = wl; b = bl;
    drow = (size_t)(r2 >> 6) * N_J + N_MED + (r2 & 63);
  }

  const float4 v = ((const float4*)src)[tid];
  float s = v.x + v.y + v.z + v.w;
  s = block_reduce_sum(s, red);
  const float mu = s * (1.0f / 1024.0f);
  const float dx0 = v.x - mu, dx1 = v.y - mu, dx2 = v.z - mu, dx3 = v.w - mu;
  float q = dx0 * dx0 + dx1 * dx1 + dx2 * dx2 + dx3 * dx3;
  q = block_reduce_sum(q, red);
  const float rstd = rsqrtf(q * (1.0f / 1024.0f) + 1e-5f);

  const float4 wv = ((const float4*)w)[tid];
  const float4 bv = ((const float4*)b)[tid];
  _Float16* dst = nin + drow * DIMC + tid * 4;
  dst[0] = (_Float16)(dx0 * rstd * wv.x + bv.x);
  dst[1] = (_Float16)(dx1 * rstd * wv.y + bv.y);
  dst[2] = (_Float16)(dx2 * rstd * wv.z + bv.z);
  dst[3] = (_Float16)(dx3 * rstd * wv.w + bv.w);
}

// ---------------------------------------------------------------------------
// Kernel 2: weight transpose + f32->f16   in[K][N] -> out[N][K]
// ---------------------------------------------------------------------------
__global__ __launch_bounds__(256) void wconv_kernel(const float* __restrict__ in,
                                                    _Float16* __restrict__ out,
                                                    int K, int N) {
  const int idx = blockIdx.x * 256 + threadIdx.x;
  if (idx < K * N) {
    const int k = idx / N, n = idx % N;
    out[(size_t)n * K + k] = (_Float16)in[idx];
  }
}

// ---------------------------------------------------------------------------
// Kernel 3: 128x128-tile WMMA GEMM, double-buffered async global->LDS staging.
// C[M,N] = A[M,K] * Bt[N,K]^T
// MODE 0: kv projection  (A=nin rows, scatter k / transposed v)
// MODE 1: q  projection  (A=latent rows of nin, store q*scale)
// MODE 2: out projection (A=z rows, store f32 d_out)
// ---------------------------------------------------------------------------
#define LDT 40  // padded LDS stride (halves), multiple of 8, conflict-free

template <int MODE>
__global__ __launch_bounds__(256) void gemm_wmma(
    const _Float16* __restrict__ A, const _Float16* __restrict__ Bt,
    _Float16* __restrict__ out_k, _Float16* __restrict__ out_v,
    _Float16* __restrict__ out_q, float* __restrict__ out_f, int Kdim) {
  __shared__ alignas(16) _Float16 lds_a[2][128 * LDT];
  __shared__ alignas(16) _Float16 lds_b[2][128 * LDT];

  const int tid = threadIdx.x;
  const int lane = tid & 31;
  const int wave = tid >> 5;
  const int wm = (wave >> 1) * 32;  // 0,32,64,96
  const int wn = (wave & 1) * 64;   // 0,64
  const int m0 = blockIdx.x * 128;
  const int n0 = blockIdx.y * 128;

  // staging assignment: 32B (two b128 async transfers) per thread per tile
  const int arow = tid >> 1;
  const int aseg = (tid & 1) << 4;
  size_t grow;
  if (MODE == 1) {
    const int g = m0 + arow;
    grow = (size_t)(g >> 6) * N_J + N_MED + (g & 63);
  } else {
    grow = (size_t)(m0 + arow);
  }
  const _Float16* agp = A + grow * Kdim + aseg;
  const _Float16* bgp = Bt + (size_t)(n0 + arow) * Kdim + aseg;
  const uint32_t la0 = lds_off(&lds_a[0][arow * LDT + aseg]);
  const uint32_t lb0 = lds_off(&lds_b[0][arow * LDT + aseg]);
  const uint32_t bstep = (uint32_t)(128 * LDT * sizeof(_Float16));

  v8f acc[2][4];
#pragma unroll
  for (int i = 0; i < 2; i++)
#pragma unroll
    for (int j = 0; j < 4; j++) acc[i][j] = (v8f){};

  // prologue: async-stage tile 0 into buffer 0
  async_g2l_b128(la0, agp);
  async_g2l_b128(la0 + 16, agp + 8);
  async_g2l_b128(lb0, bgp);
  async_g2l_b128(lb0 + 16, bgp + 8);

  for (int k0 = 0; k0 < Kdim; k0 += 32) {
    const int cur = (k0 >> 5) & 1;
    wait_async_all();   // this wave's async tile landed in LDS
    __syncthreads();    // all waves' tiles visible

    if (k0 + 32 < Kdim) {  // overlap next tile's DMA with this tile's WMMAs
      const uint32_t nb = (uint32_t)(cur ^ 1) * bstep;
      async_g2l_b128(la0 + nb, agp + k0 + 32);
      async_g2l_b128(la0 + nb + 16, agp + k0 + 40);
      async_g2l_b128(lb0 + nb, bgp + k0 + 32);
      async_g2l_b128(lb0 + nb + 16, bgp + k0 + 40);
    }

    const _Float16* la = lds_a[cur];
    const _Float16* lb = lds_b[cur];
    const v16h af0 = frag_a_ld(&la[(wm)*LDT], LDT);
    const v16h af1 = frag_a_ld(&la[(wm + 16) * LDT], LDT);
#pragma unroll
    for (int j = 0; j < 4; j++) {
      const v16h bf = frag_b_ld(&lb[(wn + j * 16) * LDT], LDT);
      acc[0][j] = wmma16(af0, bf, acc[0][j]);
      acc[1][j] = wmma16(af1, bf, acc[1][j]);
    }
    // no trailing barrier: fragment ds_loads complete (dscnt waited) before
    // their WMMA consumers, hence before the next iteration's barrier.
  }

  // epilogue
#pragma unroll
  for (int i = 0; i < 2; i++) {
#pragma unroll
    for (int j = 0; j < 4; j++) {
#pragma unroll
      for (int r = 0; r < 8; r++) {
        const int m = m0 + wm + i * 16 + ((lane >> 4) << 3) + r;
        const int n = n0 + wn + j * 16 + (lane & 15);
        const float val = acc[i][j][r];
        if (MODE == 0) {
          const int bt2 = m / N_J, jj = m % N_J;
          if (n < INNERC) {
            const int hh = n >> 6, d = n & 63;
            out_k[((size_t)(bt2 * HEADSC + hh) * N_J + jj) * DHC + d] =
                (_Float16)val;
          } else {
            const int c = n - INNERC, hh = c >> 6, d = c & 63;
            out_v[((size_t)(bt2 * HEADSC + hh) * DHC + d) * N_J + jj] =
                (_Float16)val;
          }
        } else if (MODE == 1) {
          const int bt2 = m >> 6, ii = m & 63, hh = n >> 6, d = n & 63;
          out_q[((size_t)(bt2 * HEADSC + hh) * N_LAT + ii) * DHC + d] =
              (_Float16)(val * 0.125f);  // DIM_HEAD^-0.5 folded into q
        } else {
          out_f[(size_t)m * DIMC + n] = val;
        }
      }
    }
  }
}

// ---------------------------------------------------------------------------
// Kernel 4: flash attention per (bt, head). 4 waves x 16 q-rows, j-tiles of 128.
// k/vT tiles staged with async global->LDS; out-of-range j handled by clamped
// (duplicate) loads — the softmax mask forces p=0 for j>=1088, so garbage K/V
// contributes exactly zero to max/sum/O.
// ---------------------------------------------------------------------------
#define LDK 72   // k tile LDS stride (halves)
#define LDV 136  // vT / p tile LDS stride (halves)

__global__ __launch_bounds__(128) void attn_kernel(
    const _Float16* __restrict__ q, const _Float16* __restrict__ k,
    const _Float16* __restrict__ vT, _Float16* __restrict__ z) {
  __shared__ alignas(16) _Float16 lk[128 * LDK];       // k tile  [j][d]
  __shared__ alignas(16) _Float16 lv[64 * LDV];        // vT tile [d][j]
  __shared__ alignas(16) _Float16 lp[4 * 16 * LDV];    // P strips per wave

  const int bh = blockIdx.x;        // bt*8 + h
  const int bt = bh >> 3, h = bh & 7;
  const int tid = threadIdx.x, wave = tid >> 5, lane = tid & 31;

  const _Float16* qg = q + ((size_t)bh * N_LAT + wave * 16) * DHC;
  const _Float16* kg = k + (size_t)bh * N_J * DHC;
  const _Float16* vg = vT + (size_t)bh * DHC * N_J;

  // q A-fragments straight from global (layout [i][d], ld=64)
  const v16h aq0 = frag_a_ld(qg, DHC);
  const v16h aq1 = frag_a_ld(qg + 32, DHC);

  const uint32_t lk_addr = lds_off(&lk[tid * LDK]);
  const int dr = tid >> 1, seg = (tid & 1) * 64;
  const uint32_t lv_addr = lds_off(&lv[dr * LDV + seg]);

  v8f o[4];
#pragma unroll
  for (int dt = 0; dt < 4; dt++) o[dt] = (v8f){};
  float mrow[8], lrow[8];
#pragma unroll
  for (int r = 0; r < 8; r++) { mrow[r] = -3e38f; lrow[r] = 0.0f; }

  for (int jt = 0; jt < N_J; jt += 128) {
    // ---- async-stage k tile: thread = one j row (clamped) ----
    {
      int j = jt + tid;
      if (j > N_J - 1) j = N_J - 1;
      const _Float16* src = kg + (size_t)j * DHC;
#pragma unroll
      for (int s = 0; s < 8; s++)
        async_g2l_b128(lk_addr + s * 16, src + s * 8);
    }
    // ---- async-stage vT tile: thread = half a d row (clamped chunks) ----
    {
      const _Float16* srow = vg + (size_t)dr * N_J;
#pragma unroll
      for (int s = 0; s < 8; s++) {
        int c0 = jt + seg + s * 8;
        if (c0 > N_J - 8) c0 = N_J - 8;
        async_g2l_b128(lv_addr + s * 16, srow + c0);
      }
    }
    wait_async_all();
    __syncthreads();

    // ---- S = q * k^T : 8 j-tiles, K(d)=64 -> 2 chained WMMAs each ----
    v8f s[8];
#pragma unroll
    for (int tj = 0; tj < 8; tj++) {
      const v16h b0 = frag_b_ld(&lk[(tj * 16) * LDK], LDK);
      const v16h b1 = frag_b_ld(&lk[(tj * 16) * LDK + 32], LDK);
      v8f t = wmma16(aq0, b0, (v8f){});
      s[tj] = wmma16(aq1, b1, t);
    }
    // mask out-of-range columns
#pragma unroll
    for (int tj = 0; tj < 8; tj++) {
      if (jt + tj * 16 + (lane & 15) >= N_J) {
#pragma unroll
        for (int r = 0; r < 8; r++) s[tj][r] = -3e38f;
      }
    }

    // ---- online softmax in C-layout (row == element index per half-wave) ----
    float cm[8];
#pragma unroll
    for (int r = 0; r < 8; r++) {
      cm[r] = s[0][r];
#pragma unroll
      for (int tj = 1; tj < 8; tj++) cm[r] = fmaxf(cm[r], s[tj][r]);
#pragma unroll
      for (int msk = 1; msk < 16; msk <<= 1)
        cm[r] = fmaxf(cm[r], __shfl_xor(cm[r], msk, 32));
    }
    float nm[8], alpha[8], rs[8];
#pragma unroll
    for (int r = 0; r < 8; r++) {
      nm[r] = fmaxf(mrow[r], cm[r]);
      alpha[r] = __expf(mrow[r] - nm[r]);
      rs[r] = 0.0f;
    }
    const int rbase = (lane >> 4) << 3;
#pragma unroll
    for (int tj = 0; tj < 8; tj++) {
#pragma unroll
      for (int r = 0; r < 8; r++) {
        const float p = __expf(s[tj][r] - nm[r]);
        rs[r] += p;
        lp[wave * 16 * LDV + (rbase + r) * LDV + tj * 16 + (lane & 15)] =
            (_Float16)p;
      }
    }
    asm volatile("" ::: "memory");  // order P stores before fragment reloads
#pragma unroll
    for (int r = 0; r < 8; r++) {
#pragma unroll
      for (int msk = 1; msk < 16; msk <<= 1)
        rs[r] += __shfl_xor(rs[r], msk, 32);
      lrow[r] = lrow[r] * alpha[r] + rs[r];
      mrow[r] = nm[r];
    }
#pragma unroll
    for (int dt = 0; dt < 4; dt++)
#pragma unroll
      for (int r = 0; r < 8; r++) o[dt][r] *= alpha[r];

    // ---- O += P * V ----
    v16h ap[4];
#pragma unroll
    for (int jc = 0; jc < 4; jc++)
      ap[jc] = frag_a_ld(&lp[wave * 16 * LDV + jc * 32], LDV);
#pragma unroll
    for (int dt = 0; dt < 4; dt++) {
#pragma unroll
      for (int jc = 0; jc < 4; jc++) {
        const v16h bv = frag_b_ld(&lv[(dt * 16) * LDV + jc * 32], LDV);
        o[dt] = wmma16(ap[jc], bv, o[dt]);
      }
    }
    __syncthreads();  // all reads done before next iteration's async writes
  }

  // ---- epilogue: O /= l, write z[bt*64+i][h*64+d] f16 ----
#pragma unroll
  for (int dt = 0; dt < 4; dt++) {
#pragma unroll
    for (int r = 0; r < 8; r++) {
      const int irow = wave * 16 + ((lane >> 4) << 3) + r;
      const float val = o[dt][r] / lrow[r];
      z[((size_t)(bt * N_LAT + irow)) * INNERC + h * DHC + dt * 16 +
        (lane & 15)] = (_Float16)val;
    }
  }
}

// ---------------------------------------------------------------------------
// Host launcher
// ---------------------------------------------------------------------------
extern "C" void kernel_launch(void* const* d_in, const int* in_sizes, int n_in,
                              void* d_out, int out_size, void* d_ws,
                              size_t ws_size, hipStream_t stream) {
  const float* x    = (const float*)d_in[0];
  const float* lat  = (const float*)d_in[1];
  const float* lnmw = (const float*)d_in[2];
  const float* lnmb = (const float*)d_in[3];
  const float* lnlw = (const float*)d_in[4];
  const float* lnlb = (const float*)d_in[5];
  const float* Wq   = (const float*)d_in[6];
  const float* Wkv  = (const float*)d_in[7];
  const float* Wout = (const float*)d_in[8];
  float* out = (float*)d_out;

  char* ws = (char*)d_ws;
  _Float16* nin   = (_Float16*)(ws);                         // 69632*1024 f16
  _Float16* wqT   = (_Float16*)(ws + 142606336ull);          // [512][1024]
  _Float16* wkvT  = (_Float16*)(ws + 143654912ull);          // [1024][1024]
  _Float16* woutT = (_Float16*)(ws + 145752064ull);          // [1024][512]
  _Float16* qb    = (_Float16*)(ws + 146800640ull);          // [bh][64][64]
  _Float16* kb    = (_Float16*)(ws + 150994944ull);          // [bh][1088][64]
  _Float16* vTb   = (_Float16*)(ws + 222298112ull);          // [bh][64][1088]
  _Float16* zb    = (_Float16*)(ws + 293601280ull);          // [4096][512]

  // 1) fused layernorm -> nin (media rows then latent rows per bt slice)
  ln_kernel<<<N_BT * N_MED + N_BT * N_LAT, 256, 0, stream>>>(
      x, lat, lnmw, lnmb, lnlw, lnlb, nin);

  // 2) weight transpose-convert to f16 [N][K]
  wconv_kernel<<<(1024 * 512 + 255) / 256, 256, 0, stream>>>(Wq, wqT, 1024, 512);
  wconv_kernel<<<(1024 * 1024 + 255) / 256, 256, 0, stream>>>(Wkv, wkvT, 1024, 1024);
  wconv_kernel<<<(512 * 1024 + 255) / 256, 256, 0, stream>>>(Wout, woutT, 512, 1024);

  // 3) kv projection: [69632,1024] x [1024,1024] -> k / vT (the 146 GFLOP GEMM)
  gemm_wmma<0><<<dim3(544, 8), 256, 0, stream>>>(nin, wkvT, kb, vTb, nullptr,
                                                 nullptr, 1024);
  // 4) q projection: [4096,1024] x [1024,512] -> q (scaled by 1/8)
  gemm_wmma<1><<<dim3(32, 4), 256, 0, stream>>>(nin, wqT, nullptr, nullptr, qb,
                                                nullptr, 1024);
  // 5) flash attention per (bt, head)
  attn_kernel<<<N_BT * HEADSC, 128, 0, stream>>>(qb, kb, vTb, zb);

  // 6) out projection: [4096,512] x [512,1024] -> f32 d_out
  gemm_wmma<2><<<dim3(32, 8), 256, 0, stream>>>(zb, woutT, nullptr, nullptr,
                                                nullptr, out, 512);
}
